// MultiHeadAttention_48387101557043
// MI455X (gfx1250) — matrix operand order repaired and checked by reading, not compiled, not run on hardware
//
#include <hip/hip_runtime.h>

#define EMBED 512
#define HEADS 8
#define HDIM 64
#define SEQ 2048
#define NBATCH 4
#define QTILE 128
#define KTILE 32
#define KSTR 72   // Klds row stride in ushorts (32 dw data + 4 dw TDM pad)
#define VSTR 56   // VTlds row stride in ushorts (16 dw data + 12 dw TDM pad)
#define PSTR 56   // Plds row stride
#define KBUF (KTILE * KSTR)   // one K buffer, ushorts
#define VBUF (HDIM * VSTR)    // one V^T buffer, ushorts

typedef __attribute__((ext_vector_type(16))) __bf16 v16bf;
typedef __attribute__((ext_vector_type(8)))  float  v8f;
typedef __attribute__((ext_vector_type(4)))  unsigned int u32x4;
typedef __attribute__((ext_vector_type(4)))  int i32x4;
typedef __attribute__((ext_vector_type(8)))  int i32x8;

union BF16x16 {
  v16bf v;
  uint4 q[2];
};

__device__ __forceinline__ unsigned short f2bf(float f) {
  unsigned int u = __float_as_uint(f);
  u += 0x7FFFu + ((u >> 16) & 1u);   // round-to-nearest-even
  return (unsigned short)(u >> 16);
}
__device__ __forceinline__ unsigned int f2bf2(float a, float b) {
  return (unsigned int)f2bf(a) | ((unsigned int)f2bf(b) << 16);
}
__device__ __forceinline__ uint4 pack8(float4 a, float4 b) {
  return make_uint4(f2bf2(a.x, a.y), f2bf2(a.z, a.w), f2bf2(b.x, b.y), f2bf2(b.z, b.w));
}
__device__ __forceinline__ v8f wmma_bf(v16bf a, v16bf b, v8f c) {
  return __builtin_amdgcn_wmma_f32_16x16x32_bf16(false, a, false, b, (short)0, c,
                                                 false, false);
}
// B operand / contiguous-K loads: 16 bf16 = two b128 at +0,+8 ushorts
__device__ __forceinline__ v16bf ld16_contig(const unsigned short* p) {
  BF16x16 t;
  t.q[0] = *(const uint4*)(p);
  t.q[1] = *(const uint4*)(p + 8);
  return t.v;
}
// A operand loads: K pattern {0..7,16..23}: two b128 at +0,+16 ushorts
__device__ __forceinline__ v16bf ld16_split(const unsigned short* p) {
  BF16x16 t;
  t.q[0] = *(const uint4*)(p);
  t.q[1] = *(const uint4*)(p + 16);
  return t.v;
}

// ---------------------------------------------------------------------------
// Tensor Data Mover: 2D bf16 tile (tile_w x tile_h) from strided global tensor
// into LDS, with hardware row padding producing our bank-conflict-free strides.
// D# layout per CDNA5 ISA.
// ---------------------------------------------------------------------------
__device__ __forceinline__ void tdm_load_2d_bf16(unsigned lds_addr, const void* gaddr,
                                                 unsigned tile_w, unsigned tile_h,
                                                 unsigned stride_elems,
                                                 unsigned pad_interval_code,
                                                 unsigned pad_amount_code) {
  unsigned long long ga = (unsigned long long)(size_t)gaddr;
  u32x4 g0;
  g0[0] = 1u;                                            // count=1, no gather
  g0[1] = lds_addr;                                      // [63:32] lds byte addr
  g0[2] = (unsigned)(ga & 0xFFFFFFFFu);                  // [95:64] global addr lo
  g0[3] = (unsigned)((ga >> 32) & 0x01FFFFFFu)           // [120:96] global addr hi
        | 0x80000000u;                                   // [127:126] type=2 (image)
  i32x8 g1;
  g1[0] = (int)((1u << 16)                               // data_size = 2 bytes
              | (1u << 20)                               // pad_enable
              | (pad_interval_code << 22)
              | (pad_amount_code << 25));
  g1[1] = (int)((tile_w & 0xFFFFu) << 16);               // abar=0 | tensor_dim0.lo
  g1[2] = (int)(((tile_w >> 16) & 0xFFFFu)               // tensor_dim0.hi
              | ((tile_h & 0xFFFFu) << 16));             // tensor_dim1.lo
  g1[3] = (int)(((tile_h >> 16) & 0xFFFFu)               // tensor_dim1.hi
              | ((tile_w & 0xFFFFu) << 16));             // tile_dim0
  g1[4] = (int)(tile_h & 0xFFFFu);                       // tile_dim1 | tile_dim2=0
  g1[5] = (int)stride_elems;                             // tensor_dim0_stride.lo
  g1[6] = 0;                                             // stride.hi | dim1_stride.lo
  g1[7] = 0;                                             // dim1_stride.hi
  i32x4 z = {0, 0, 0, 0};
#if __clang_major__ >= 23
  i32x8 z8 = {0, 0, 0, 0, 0, 0, 0, 0};
  __builtin_amdgcn_tensor_load_to_lds(g0, g1, z, z, z8, 0);
#else
  __builtin_amdgcn_tensor_load_to_lds(g0, g1, z, z, 0);
#endif
}

// ---------------------------------------------------------------------------
// Prep: K fp32 -> bf16 row-major; V fp32 -> bf16 transposed per head
// VT layout: [n][h][d(64)][s(2048)]
// ---------------------------------------------------------------------------
__global__ __launch_bounds__(256) void prep_kernel(
    const float* __restrict__ K, const float* __restrict__ V,
    unsigned short* __restrict__ Kbf, unsigned short* __restrict__ VT) {
  const int i = blockIdx.x * 256 + threadIdx.x;   // over NBATCH*SEQ*EMBED
  Kbf[i] = f2bf(K[i]);
  const int e = i & (EMBED - 1);
  const int s = (i >> 9) & (SEQ - 1);
  const int n = i >> 20;
  const int h = e >> 6, d = e & (HDIM - 1);
  VT[(((size_t)(n * HEADS + h) * HDIM + d) * SEQ) + s] = f2bf(V[i]);
}

__global__ __launch_bounds__(256) void wconv_kernel(const float* __restrict__ W,
                                                    unsigned short* __restrict__ Wbf) {
  const int i = blockIdx.x * 256 + threadIdx.x;
  Wbf[i] = f2bf(W[i]);
}

// ---------------------------------------------------------------------------
// Kernel 1: flash attention.  Block = (n, h, 128-row q tile), 8 waves.
// Wave 0 streams K / V^T tiles with the Tensor Data Mover into double-buffered
// LDS (issue tile i+1, wait tensorcnt<=2 for tile i, barrier, compute) so the
// DMA overlaps the 9 WMMAs per 32-key iteration (4 QK^T + 1 row-sum + 4 PV).
// ---------------------------------------------------------------------------
__global__ __launch_bounds__(256) void attn_fa_kernel(
    const unsigned short* __restrict__ Kbf, const unsigned short* __restrict__ VT,
    const float* __restrict__ Qg, const int* __restrict__ maskg,
    unsigned short* __restrict__ Xws) {
  __shared__ unsigned short Klds[2 * KBUF];         // K tiles, row-major bf16
  __shared__ unsigned short VTlds[2 * VBUF];        // V^T tiles, bf16
  __shared__ unsigned short Plds[8 * 16 * PSTR];    // per-wave P scratch

  const int b  = blockIdx.x;
  const int qt = b & 15;
  const int h  = (b >> 4) & 7;
  const int n  = b >> 7;

  const int tid  = threadIdx.x;
  const int wave = tid >> 5;
  const int lane = tid & 31;
  const int hi   = lane >> 4;
  const int ln   = lane & 15;

  // ---- Q strip (16 x 64 fp32 -> bf16) as two A operands, K chunks 0-31 / 32-63
  const int qrow = qt * QTILE + wave * 16 + ln;
  const float* qp = Qg + ((size_t)(n * SEQ + qrow)) * EMBED + h * HDIM;
  BF16x16 qa0, qa1;
  {
    const int b0 = hi ? 16 : 0;                // A layout: lanes>=16 hold K 16..31
    float4 f0 = *(const float4*)(qp + b0);
    float4 f1 = *(const float4*)(qp + b0 + 4);
    float4 f2 = *(const float4*)(qp + b0 + 8);
    float4 f3 = *(const float4*)(qp + b0 + 12);
    qa0.q[0] = pack8(f0, f1);
    qa0.q[1] = pack8(f2, f3);
    f0 = *(const float4*)(qp + 32 + b0);
    f1 = *(const float4*)(qp + 32 + b0 + 4);
    f2 = *(const float4*)(qp + 32 + b0 + 8);
    f3 = *(const float4*)(qp + 32 + b0 + 12);
    qa1.q[0] = pack8(f0, f1);
    qa1.q[1] = pack8(f2, f3);
  }

  BF16x16 ones;                                     // B = 1s: row-sum via WMMA
  ones.q[0] = make_uint4(0x3F803F80u, 0x3F803F80u, 0x3F803F80u, 0x3F803F80u);
  ones.q[1] = ones.q[0];

  v8f acc0 = {}, acc1 = {}, acc2 = {}, acc3 = {};   // O tile, dims 0-15/.../48-63
  float mrow[8], lrow[8];
#pragma unroll
  for (int r = 0; r < 8; ++r) { mrow[r] = -1e30f; lrow[r] = 0.0f; }

  unsigned short* pb = Plds + wave * 16 * PSTR;
  const unsigned kldsa = (unsigned)(size_t)(void*)Klds;
  const unsigned vldsa = (unsigned)(size_t)(void*)VTlds;
  const unsigned short* ktile0 = Kbf + ((size_t)(n * SEQ)) * EMBED + h * HDIM;
  const unsigned short* vtile0 = VT + ((size_t)(n * HEADS + h) * HDIM) * SEQ;
  const float SCALE = 0.044194173824159216f;        // 1/sqrt(512) per reference

  // ---- prologue: start DMA of tile 0 into buffer 0 -------------------------
  if (tid < 32) {
    tdm_load_2d_bf16(kldsa, ktile0, HDIM, KTILE, EMBED, 4, 3);
    tdm_load_2d_bf16(vldsa, vtile0, KTILE, HDIM, SEQ, 3, 11);
  }

  for (int jj = 0; jj < SEQ; jj += KTILE) {
    const int cur = (jj >> 5) & 1;
    const unsigned short* kb = Klds + cur * KBUF + hi * 16;
    const unsigned short* vb = VTlds + cur * VBUF + hi * 16;

    if (tid < 32) {
      if (jj + KTILE < SEQ) {
        // issue next tile into the other buffer (fully consumed one iter ago)
        const unsigned nk = kldsa + (unsigned)((cur ^ 1) * KBUF * 2);
        const unsigned nv = vldsa + (unsigned)((cur ^ 1) * VBUF * 2);
        tdm_load_2d_bf16(nk, ktile0 + (size_t)(jj + KTILE) * EMBED, HDIM, KTILE,
                         EMBED, 4, 3);
        tdm_load_2d_bf16(nv, vtile0 + jj + KTILE, KTILE, HDIM, SEQ, 3, 11);
        __builtin_amdgcn_s_wait_tensorcnt(2);     // in-order: current tile landed
      } else {
        __builtin_amdgcn_s_wait_tensorcnt(0);
      }
    }
    __syncthreads();                              // current tile visible to all

    // ---- S = Q K^T : 16q x 32k, f32 accum ---------------------------------
    v8f s0 = {}, s1 = {};
    s0 = wmma_bf(qa0.v, ld16_contig(kb + (0  + ln) * KSTR + 0),  s0);
    s0 = wmma_bf(qa1.v, ld16_contig(kb + (0  + ln) * KSTR + 32), s0);
    s1 = wmma_bf(qa0.v, ld16_contig(kb + (16 + ln) * KSTR + 0),  s1);
    s1 = wmma_bf(qa1.v, ld16_contig(kb + (16 + ln) * KSTR + 32), s1);

    // ---- mask + scale + online softmax (max via butterflies) --------------
    const int mk0 = maskg[n * SEQ + jj + ln];
    const int mk1 = maskg[n * SEQ + jj + 16 + ln];
#pragma unroll
    for (int r = 0; r < 8; ++r) {
      float a = mk0 ? s0[r] * SCALE : -1e30f;
      float c = mk1 ? s1[r] * SCALE : -1e30f;
      float mx = fmaxf(a, c);                     // row lives in one 16-lane half
      mx = fmaxf(mx, __shfl_xor(mx, 1, 32));
      mx = fmaxf(mx, __shfl_xor(mx, 2, 32));
      mx = fmaxf(mx, __shfl_xor(mx, 4, 32));
      mx = fmaxf(mx, __shfl_xor(mx, 8, 32));
      float mnew = fmaxf(mrow[r], mx);
      float corr = __expf(mrow[r] - mnew);
      float p0 = __expf(a - mnew);
      float p1 = __expf(c - mnew);
      lrow[r] *= corr;
      mrow[r] = mnew;
      acc0[r] *= corr; acc1[r] *= corr; acc2[r] *= corr; acc3[r] *= corr;
      const int prow = r + 8 * hi;                // D-layout -> row-major LDS
      pb[prow * PSTR + ln]      = f2bf(p0);
      pb[prow * PSTR + 16 + ln] = f2bf(p1);
    }

    // ---- P in A layout; row sums via P x 1s; O += P V ---------------------
    v16bf pav = ld16_contig(pb + ln * PSTR + (hi ? 16 : 0));
    v8f zero = {};
    v8f lsum = wmma_bf(pav, ones.v, zero);
#pragma unroll
    for (int r = 0; r < 8; ++r) lrow[r] += lsum[r];
    acc0 = wmma_bf(pav, ld16_contig(vb + (0  + ln) * VSTR), acc0);
    acc1 = wmma_bf(pav, ld16_contig(vb + (16 + ln) * VSTR), acc1);
    acc2 = wmma_bf(pav, ld16_contig(vb + (32 + ln) * VSTR), acc2);
    acc3 = wmma_bf(pav, ld16_contig(vb + (48 + ln) * VSTR), acc3);

    __syncthreads();                              // buffer free for next DMA
  }

  // ---- normalize, store attention output X as bf16 -------------------------
#pragma unroll
  for (int r = 0; r < 8; ++r) {
    const float inv = lrow[r] > 0.0f ? 1.0f / lrow[r] : 0.0f;
    const int row = n * SEQ + qt * QTILE + wave * 16 + r + 8 * hi;
    unsigned short* xp = Xws + (size_t)row * EMBED + h * HDIM + ln;
    xp[0]  = f2bf(acc0[r] * inv);
    xp[16] = f2bf(acc1[r] * inv);
    xp[32] = f2bf(acc2[r] * inv);
    xp[48] = f2bf(acc3[r] * inv);
  }
}

// ---------------------------------------------------------------------------
// Kernel 2: Y = X @ W^T + b.  Block = 128 rows x 64 cols, wave = 16x64 tile.
// ---------------------------------------------------------------------------
__global__ __launch_bounds__(256) void proj_kernel(
    const unsigned short* __restrict__ X, const unsigned short* __restrict__ Wb,
    const float* __restrict__ bias, float* __restrict__ out) {
  const int wave = threadIdx.x >> 5, lane = threadIdx.x & 31;
  const int hi = lane >> 4, ln = lane & 15;
  const int rbase = blockIdx.x * 128 + wave * 16;
  const int obase = blockIdx.y * 64;

  v8f c0 = {}, c1 = {}, c2 = {}, c3 = {};
  const unsigned short* xa = X + (size_t)(rbase + ln) * EMBED + (hi ? 16 : 0);
  const unsigned short* wb = Wb + hi * 16;          // B: lanes>=16 hold K 16..31
#pragma unroll 4
  for (int kk = 0; kk < EMBED; kk += 32) {
    v16bf a = ld16_contig(xa + kk);                 // A: X rows, bf16, direct layout
    c0 = wmma_bf(a, ld16_contig(wb + (size_t)(obase + 0  + ln) * EMBED + kk), c0);
    c1 = wmma_bf(a, ld16_contig(wb + (size_t)(obase + 16 + ln) * EMBED + kk), c1);
    c2 = wmma_bf(a, ld16_contig(wb + (size_t)(obase + 32 + ln) * EMBED + kk), c2);
    c3 = wmma_bf(a, ld16_contig(wb + (size_t)(obase + 48 + ln) * EMBED + kk), c3);
  }
  const float b0 = bias[obase + 0  + ln];
  const float b1 = bias[obase + 16 + ln];
  const float b2 = bias[obase + 32 + ln];
  const float b3 = bias[obase + 48 + ln];
#pragma unroll
  for (int r = 0; r < 8; ++r) {
    const int row = rbase + r + 8 * hi;
    float* op = out + (size_t)row * EMBED + obase + ln;
    op[0]  = c0[r] + b0;
    op[16] = c1[r] + b1;
    op[32] = c2[r] + b2;
    op[48] = c3[r] + b3;
  }
}

extern "C" void kernel_launch(void* const* d_in, const int* in_sizes, int n_in,
                              void* d_out, int out_size, void* d_ws, size_t ws_size,
                              hipStream_t stream) {
  (void)in_sizes; (void)n_in; (void)out_size; (void)ws_size;
  const float* Vg = (const float*)d_in[0];
  const float* Kg = (const float*)d_in[1];
  const float* Qg = (const float*)d_in[2];
  const int*   Mg = (const int*)d_in[3];
  const float* Wg = (const float*)d_in[4];
  const float* Bg = (const float*)d_in[5];
  float* out = (float*)d_out;

  // workspace: X bf16 (8MB) | W bf16 (0.5MB) | K bf16 (8MB) | V^T bf16 (8MB)
  const size_t XB = (size_t)NBATCH * SEQ * EMBED * sizeof(unsigned short);
  const size_t WB = (size_t)EMBED * EMBED * sizeof(unsigned short);
  unsigned short* Xws = (unsigned short*)d_ws;
  unsigned short* Wbf = (unsigned short*)((char*)d_ws + XB);
  unsigned short* Kbf = (unsigned short*)((char*)d_ws + XB + WB);
  unsigned short* VTw = (unsigned short*)((char*)d_ws + XB + WB + XB);

  wconv_kernel<<<(EMBED * EMBED) / 256, 256, 0, stream>>>(Wg, Wbf);
  prep_kernel<<<(NBATCH * SEQ * EMBED) / 256, 256, 0, stream>>>(Kg, Vg, Kbf, VTw);
  attn_fa_kernel<<<NBATCH * HEADS * (SEQ / QTILE), 256, 0, stream>>>(Kbf, VTw, Qg, Mg, Xws);
  proj_kernel<<<dim3((NBATCH * SEQ) / 128, EMBED / 64), 256, 0, stream>>>(Xws, Wbf, Bg, out);
}
